// SCSWMHSA_66228395705132
// MI455X (gfx1250) — matrix operand order repaired and checked by reading, not compiled, not run on hardware
//
#include <hip/hip_runtime.h>

// CSWin windowed MHSA + LEPE for gfx1250 (MI455X), wave32 + v_wmma_f32_16x16x32_f16.
// One block per (window, head): Q,K,V are 256x32. HBM-bound (~134MB @ 23.3TB/s).
// v3: S computed TRANSPOSED (S' = K·Q^T) so the C-fragment of S' IS the A-fragment
//     of P for the P·V GEMM (in-lane f32->f16 only) -> no LDS re-striping.
// v4: __syncthreads() per pass keeps allocation under VGPR 256 (no s_set_vgpr_msb).
// v5: fully-unrolled gather (max MLP on the HBM-bound phase) + double-buffered
//     K A-fragments so pre-WMMA waits don't drain DScnt to zero.

typedef _Float16 half_t;
typedef __attribute__((ext_vector_type(16))) _Float16 v16h;
typedef __attribute__((ext_vector_type(8)))  float    v8f;

#define NC    256    // channels
#define ND    32     // head dim
#define HW    4096   // H*W = 64*64
#define NL    256    // tokens per window (64 x 4)
#define VT_S  264    // Vt row stride in halves (pad, 16B-aligned)
#define SCALE_F 0.17677669529663687f  // 32^-0.5

__device__ __forceinline__ v16h ld16(const half_t* p) {
  // 16 consecutive halves (two 16B LDS loads)
  v16h r;
  *(uint4*)&r       = *(const uint4*)p;
  *((uint4*)&r + 1) = *(const uint4*)(p + 8);
  return r;
}
__device__ __forceinline__ v16h ld8_8(const half_t* p0, const half_t* p1) {
  // A-fragment: halves 0..7 from p0, 8..15 from p1
  v16h r;
  *(uint4*)&r       = *(const uint4*)p0;
  *((uint4*)&r + 1) = *(const uint4*)p1;
  return r;
}

__global__ __launch_bounds__(256) void cswin_attn_kernel(
    const float* __restrict__ temp,   // [B,3,C,H,W] f32
    const float* __restrict__ wgt,    // [C,1,3,3]   f32
    const float* __restrict__ bias,   // [C]         f32
    float*       __restrict__ out)    // [B,C,H,W] flat f32
{
  __shared__ half_t Qs[NL * ND];      // 16 KB   Q (pre-scaled), row-major [tok][d]
  __shared__ half_t Ks[NL * ND];      // 16 KB   K,              row-major [tok][d]
  __shared__ half_t Vt[ND * VT_S];    // 16.5 KB V transposed    [d][tok]

  const int blk  = blockIdx.x;        // 0..1023
  const int head = blk & 7;
  const int win  = blk >> 3;          // 0..127
  const int b    = win >> 4;
  const int ww   = win & 15;

  const int tid  = threadIdx.x;
  const int lane = tid & 31;
  const int wave = tid >> 5;
  const int hup  = lane >> 4;         // lane half (0: lanes 0-15, 1: lanes 16-31)
  const int l16  = lane & 15;

  // ---- gather Q,K,V for this (window, head) into LDS, f32 -> f16 ----
  // Fully unrolled: queue all 96 scattered dword loads for max memory-level
  // parallelism (this is the HBM-bound phase; register pressure here is low).
  {
    const int tok = tid;                                   // one token per thread
    const int sp  = (tok >> 2) * 64 + (tok & 3) * 16 + ww; // h = tok/4, w = (tok%4)*16+ww
    const float* pQ = temp + ((size_t)(b * 3 + 0) * NC + head * ND) * HW + sp;
    const float* pK = temp + ((size_t)(b * 3 + 1) * NC + head * ND) * HW + sp;
    const float* pV = temp + ((size_t)(b * 3 + 2) * NC + head * ND) * HW + sp;
#pragma unroll
    for (int d = 0; d < ND; ++d) {
      float q = pQ[(size_t)d * HW];
      float k = pK[(size_t)d * HW];
      float v = pV[(size_t)d * HW];
      Qs[tok * ND + d]   = (half_t)(q * SCALE_F);
      Ks[tok * ND + d]   = (half_t)k;
      Vt[d * VT_S + tok] = (half_t)v;
    }
  }

  // depthwise 3x3 weights/bias for this lane's two channels (d = l16, l16+16)
  float w9a[9], w9b[9];
  const int ca = head * ND + l16;
  const int cb = ca + 16;
#pragma unroll
  for (int i = 0; i < 9; ++i) { w9a[i] = wgt[ca * 9 + i]; w9b[i] = wgt[cb * 9 + i]; }
  const float bA = bias[ca], bB = bias[cb];

  // Each wave owns q-tiles qt = wave and wave+8 (16 query tokens each).
  for (int pass = 0; pass < 2; ++pass) {
    // Barrier: (a) pass 0: gather visible to all waves; (b) acts as an LDS clobber so
    // the 16 K A-fragments are re-loaded per pass instead of staying live (keeps the
    // allocation under VGPR 256 -> no s_set_vgpr_msb mode).
    __syncthreads();

    const int qt = wave + pass * 8;

    // B fragment of Q for this q-tile: B[d][q], lane = q, halves = d group per lane half
    const int qtok = qt * 16 + l16;
    const v16h bq = ld8_8(&Qs[qtok * ND + hup * 8], &Qs[qtok * ND + 16 + hup * 8]);

    // S' = K @ Q^T : 16 tiles along k (M dim), one WMMA each (K-dim = 32 = head dim).
    // Double-buffer the K A-fragment so each WMMA waits on the *previous* load pair.
    v8f acc[16];
    const half_t* krow0 = &Ks[l16 * ND];
    v16h ak = ld8_8(krow0 + hup * 8, krow0 + 16 + hup * 8);
#pragma unroll
    for (int mt = 0; mt < 16; ++mt) {
      v16h akn;
      if (mt < 15) {
        const half_t* krow = &Ks[((mt + 1) * 16 + l16) * ND];
        akn = ld8_8(krow + hup * 8, krow + 16 + hup * 8);
      }
      v8f cz = {0.f,0.f,0.f,0.f,0.f,0.f,0.f,0.f};
      acc[mt] = __builtin_amdgcn_wmma_f32_16x16x32_f16(
          false, ak, false, bq, (short)0, cz, false, false);
      if (mt < 15) ak = akn;
    }

    // ---- softmax over k for fixed q = l16. Lane holds 128 of 256 k-values of its row;
    //      the other 128 live in the partner lane (xor 16).
    float m = acc[0][0];
#pragma unroll
    for (int mt = 0; mt < 16; ++mt)
#pragma unroll
      for (int r = 0; r < 8; ++r) m = fmaxf(m, acc[mt][r]);
    m = fmaxf(m, __shfl_xor(m, 16, 32));

    float s = 0.f;
#pragma unroll
    for (int mt = 0; mt < 16; ++mt)
#pragma unroll
      for (int r = 0; r < 8; ++r) {
        const float e = __expf(acc[mt][r] - m);
        acc[mt][r] = e;
        s += e;
      }
    s += __shfl_xor(s, 16, 32);
    const float rinv = 1.0f / s;

    // ---- O = P @ V : A fragment of P is exactly acc[2kt], acc[2kt+1] (in-lane cvt)
    v8f c0 = {0.f,0.f,0.f,0.f,0.f,0.f,0.f,0.f};
    v8f c1 = c0;
#pragma unroll
    for (int kt = 0; kt < 8; ++kt) {
      v16h as;
#pragma unroll
      for (int i = 0; i < 8; ++i) {
        as[i]     = (half_t)(acc[2 * kt][i]     * rinv);
        as[8 + i] = (half_t)(acc[2 * kt + 1][i] * rinv);
      }
      const int  kb  = kt * 32 + hup * 8;
      const v16h bv0 = ld8_8(&Vt[(l16)      * VT_S + kb], &Vt[(l16)      * VT_S + kb + 16]); // B-frag: N = dim, K = token
      const v16h bv1 = ld8_8(&Vt[(16 + l16) * VT_S + kb], &Vt[(16 + l16) * VT_S + kb + 16]);
      c0 = __builtin_amdgcn_wmma_f32_16x16x32_f16(false, as, false, bv0, (short)0, c0, false, false);
      c1 = __builtin_amdgcn_wmma_f32_16x16x32_f16(false, as, false, bv1, (short)0, c1, false, false);
    }

    // ---- epilogue: LEPE (depthwise 3x3, zero-pad on 64x4 window) + store NCHW
#pragma unroll
    for (int r = 0; r < 8; ++r) {
      const int mrow = r + hup * 8;
      const int tok  = qt * 16 + mrow;
      const int hs = tok >> 2, ws = tok & 3;
      float l0 = bA, l1 = bB;
#pragma unroll
      for (int i = -1; i <= 1; ++i) {
#pragma unroll
        for (int j = -1; j <= 1; ++j) {
          const int h2 = hs + i, w2 = ws + j;
          const bool ok = ((unsigned)h2 < 64u) & ((unsigned)w2 < 4u);
          const int t2 = h2 * 4 + w2;
          const float va = ok ? (float)Vt[l16 * VT_S + t2]        : 0.f;
          const float vb = ok ? (float)Vt[(16 + l16) * VT_S + t2] : 0.f;
          const int wi = (i + 1) * 3 + (j + 1);
          l0 += w9a[wi] * va;
          l1 += w9b[wi] * vb;
        }
      }
      const size_t o = ((size_t)b * NC + head * ND) * HW + hs * 64 + ws * 16 + ww;
      out[o + (size_t)l16        * HW] = c0[r] + l0;
      out[o + (size_t)(16 + l16) * HW] = c1[r] + l1;
    }
  }
}

extern "C" void kernel_launch(void* const* d_in, const int* in_sizes, int n_in,
                              void* d_out, int out_size, void* d_ws, size_t ws_size,
                              hipStream_t stream) {
  const float* temp = (const float*)d_in[0];  // [8,3,256,64,64]
  const float* wgt  = (const float*)d_in[1];  // [256,1,3,3]
  const float* bias = (const float*)d_in[2];  // [256]
  float* out = (float*)d_out;                 // 8*4096*256 f32

  dim3 grid(128 * 8);   // (window, head)
  dim3 block(256);      // 8 waves
  hipLaunchKernelGGL(cswin_attn_kernel, grid, block, 0, stream, temp, wgt, bias, out);
}
